// Futoshiki_BIN_CNN_16123307229948
// MI455X (gfx1250) — matrix operand order repaired and checked by reading, not compiled, not run on hardware
//
#include <hip/hip_runtime.h>
#include <hip/hip_bf16.h>

typedef _Float16 v16h __attribute__((ext_vector_type(16)));
typedef float    v8f  __attribute__((ext_vector_type(8)));

union AFrag { v16h v; unsigned int u[8]; uint4 q[2]; };

__device__ __forceinline__ v8f wmma_f16(v16h a, v16h b, v8f c) {
  // D = A(16x32 f16) * B(32x16 f16) + C(16x16 f32)
  return __builtin_amdgcn_wmma_f32_16x16x32_f16(false, a, false, b, (short)0, c, false, false);
}

// 16-bit A-matrix 16x32 per-lane K mapping (ISA 7.12.2): lane group g = lane>>4.
// Per lane the 16 halves are two contiguous K-runs: [8g, 8g+8) and [16+8g, 16+8g+8).
__device__ __forceinline__ int kmapA(int j, int g) {
  return (j < 8) ? (j + 8 * g) : (j + 8 + 8 * g);
}

__device__ __forceinline__ unsigned int pack2(float a, float b) {
  union { _Float16 h[2]; unsigned int u; } p;
  p.h[0] = (_Float16)a; p.h[1] = (_Float16)b;
  return p.u;
}

__device__ __forceinline__ float sigmoidf(float x) { return 1.f / (1.f + __expf(-x)); }

// ---------------------------------------------------------------------------
// Weight pre-swizzle: store B-matrix fragments exactly in the per-lane VGPR
// layout of v_wmma_f32_16x16x32_f16:  B element (K = kc*32 + kmapA(j, lane>>4),
// N = nt*16 + (lane&15)).  Fragment = 512 halves; per-lane slice = 16 halves,
// contiguous -> hot-loop fragment load is one 32-byte LDS/L2 load.
// MLP frag buffer per edge type: [L1: 5kc x 4nt][L2: 2x4][L3: 2x4][L4: 2x4]
//   = 44 frags = 22528 halves.
// ---------------------------------------------------------------------------
__global__ __launch_bounds__(256) void prep_mlp_frag(
    const float* __restrict__ w1, const float* __restrict__ w2,
    const float* __restrict__ w3, const float* __restrict__ w4,
    _Float16* __restrict__ out)
{
  const int tid = blockIdx.x * 256 + threadIdx.x;
  if (tid >= 22528) return;
  const int fragIdx = tid >> 9;
  const int within  = tid & 511;
  const int lane = within >> 4;
  const int j    = within & 15;
  const int g = lane >> 4, lm = lane & 15;
  const float* w; int in_dim, kc, nt;
  if (fragIdx < 20)      { w = w1; in_dim = 144; kc = fragIdx >> 2;        nt = fragIdx & 3; }
  else if (fragIdx < 28) { w = w2; in_dim = 64;  kc = (fragIdx - 20) >> 2; nt = (fragIdx - 20) & 3; }
  else if (fragIdx < 36) { w = w3; in_dim = 64;  kc = (fragIdx - 28) >> 2; nt = (fragIdx - 28) & 3; }
  else                   { w = w4; in_dim = 64;  kc = (fragIdx - 36) >> 2; nt = (fragIdx - 36) & 3; }
  const int K = kc * 32 + kmapA(j, g);
  const int Ncol = nt * 16 + lm;
  const float v = (K < in_dim) ? w[Ncol * in_dim + K] : 0.f;
  out[tid] = (_Float16)v;
}

// LSTM: B = [K=384][N=256], K<320 from w_ih[N][K], else w_hh[N][K-320].
// 12 kc x 16 nt fragments = 98304 halves.
__global__ __launch_bounds__(256) void prep_lstm_frag(
    const float* __restrict__ w_ih, const float* __restrict__ w_hh,
    _Float16* __restrict__ out)
{
  const int tid = blockIdx.x * 256 + threadIdx.x;
  if (tid >= 98304) return;
  const int fragIdx = tid >> 9;
  const int within  = tid & 511;
  const int lane = within >> 4;
  const int j    = within & 15;
  const int g = lane >> 4, lm = lane & 15;
  const int kc = fragIdx >> 4, nt = fragIdx & 15;
  const int K = kc * 32 + kmapA(j, g);
  const int Ncol = nt * 16 + lm;
  const float v = (K < 320) ? w_ih[Ncol * 320 + K] : w_hh[Ncol * 64 + (K - 320)];
  out[tid] = (_Float16)v;
}

// ---------------------------------------------------------------------------
// State init: x = embed[cell_idx (with -1 -> 2)], h_gather = f16(x), rh = 0, c = 0
// ---------------------------------------------------------------------------
__global__ __launch_bounds__(256) void init_state(
    const int* __restrict__ cell_idx, const float* __restrict__ embed,
    float* __restrict__ xbuf, _Float16* __restrict__ hg,
    _Float16* __restrict__ rh, float* __restrict__ cbuf, int N_)
{
  const long long t = (long long)blockIdx.x * 256 + threadIdx.x;
  if (t >= (long long)N_ * 64) return;
  const int n = (int)(t >> 6), j = (int)(t & 63);
  int ci = cell_idx[n];
  if (ci == -1) ci = 2;
  const float x = embed[ci * 64 + j];
  xbuf[t] = x;
  hg[t]   = (_Float16)x;
  rh[t]   = (_Float16)0.f;
  cbuf[t] = 0.f;
}

__global__ __launch_bounds__(256) void zero_msg(float* __restrict__ msg, int N_) {
  const long long t = (long long)blockIdx.x * 256 + threadIdx.x;
  if (t < (long long)N_ * 256) msg[t] = 0.f;
}

// ---------------------------------------------------------------------------
// Edge MLP, persistent-grid: each block loads the 45KB weight image into LDS
// ONCE, then each wave grid-strides over 16-edge tiles (44 WMMAs / tile).
// A fragments are built with 2x b128 loads (per-lane K-runs are contiguous);
// inter-layer transpose goes through a per-wave 16x64 f16 LDS tile; epilogue
// gets per-row dst via wave shuffle and scatter-adds with global_atomic_add_f32.
// Tile loop condition is wave-uniform -> EXEC is all-ones at every WMMA.
// ---------------------------------------------------------------------------
__global__ __launch_bounds__(256) void edge_mlp_kernel(
    const _Float16* __restrict__ hg,     // [N,64] f16
    const int* __restrict__ edges,       // [2,E]: src = [0,E), dst = [E,2E)
    const _Float16* __restrict__ wfrag,  // 22528 halves, fragment-swizzled
    const float* __restrict__ b1, const float* __restrict__ b2,
    const float* __restrict__ b3, const float* __restrict__ b4,
    const float* __restrict__ eemb,      // 16 f32
    float* __restrict__ msgbuf,          // [N,256] f32
    int typeOff, int E_, int N_)
{
  __shared__ __align__(32) _Float16 lds[22528 + 8 * 1024]; // weights + 8 wave stage tiles

  // cooperative copy of pre-swizzled weights -> LDS (2816 x uint4), once per block
  {
    const uint4* s = (const uint4*)wfrag;
    uint4* d = (uint4*)lds;
    for (int i = threadIdx.x; i < 2816; i += 256) d[i] = s[i];
  }
  __syncthreads();

  const int wave = threadIdx.x >> 5;
  const int lane = threadIdx.x & 31;
  const int g  = lane >> 4;
  const int lm = lane & 15;
  _Float16* hstage = lds + 22528 + wave * 1024; // 16 rows x 64 cols f16

  // loop-invariant bias slices (per lane column lm within each 16-wide n-tile)
  float bb1[4], bb2[4], bb3[4], bb4[4];
  #pragma unroll
  for (int t = 0; t < 4; ++t) {
    bb1[t] = b1[t * 16 + lm];
    bb2[t] = b2[t * 16 + lm];
    bb3[t] = b3[t * 16 + lm];
    bb4[t] = b4[t * 16 + lm];
  }
  // loop-invariant edge-embedding fragment for layer-1 kc=4 (K in [128,160))
  AFrag ee;
  #pragma unroll
  for (int jj = 0; jj < 8; ++jj) {
    const int K = 128 + kmapA(2 * jj, g);
    ee.u[jj] = (K < 144) ? pack2(eemb[K - 128], eemb[K - 127]) : 0u;
  }

  const int numTiles = (E_ + 15) >> 4;
  const int tileStride = gridDim.x * 8;

  for (int tile = blockIdx.x * 8 + wave; tile < numTiles; tile += tileStride) {
    const int e0 = tile * 16;
    {
      int nt2 = e0 + tileStride * 16;
      if (nt2 >= E_) nt2 = E_ - 1;               // branchless speculative prefetch
      __builtin_prefetch(edges + nt2, 0, 0);
      __builtin_prefetch(edges + E_ + nt2, 0, 0);
    }
    const int edge = e0 + lm;
    const int eclamp = edge < E_ ? edge : (E_ - 1);
    const int src = edges[eclamp];
    const int dst = edges[E_ + eclamp];
    const _Float16* hsrc = hg + (long long)src * 64;
    const _Float16* hdst = hg + (long long)dst * 64;

    // ---- Layer 1: K = 160 (2H=128 | eemb=16 | pad=16) ----
    v8f acc[4] = {};
    #pragma unroll
    for (int kc = 0; kc < 5; ++kc) {
      AFrag a;
      if (kc < 2) {
        a.q[0] = *(const uint4*)(hsrc + kc * 32 + 8 * g);
        a.q[1] = *(const uint4*)(hsrc + kc * 32 + 16 + 8 * g);
      } else if (kc < 4) {
        a.q[0] = *(const uint4*)(hdst + (kc - 2) * 32 + 8 * g);
        a.q[1] = *(const uint4*)(hdst + (kc - 2) * 32 + 16 + 8 * g);
      } else {
        a = ee;
      }
      #pragma unroll
      for (int t = 0; t < 4; ++t) {
        const v16h b = *(const v16h*)(lds + (kc * 4 + t) * 512 + lane * 16);
        acc[t] = wmma_f16(a.v, b, acc[t]);
      }
    }
    #pragma unroll
    for (int t = 0; t < 4; ++t) {
      #pragma unroll
      for (int r = 0; r < 8; ++r)
        hstage[(r + 8 * g) * 64 + t * 16 + lm] = (_Float16)fmaxf(acc[t][r] + bb1[t], 0.f);
    }

    // ---- Layers 2 and 3: K = 64, relu ----
    #pragma unroll
    for (int layer = 0; layer < 2; ++layer) {
      const int base = (layer == 0) ? 10240 : 14336;
      AFrag a0, a1;
      a0.q[0] = *(const uint4*)(hstage + lm * 64 + 8 * g);
      a0.q[1] = *(const uint4*)(hstage + lm * 64 + 16 + 8 * g);
      a1.q[0] = *(const uint4*)(hstage + lm * 64 + 32 + 8 * g);
      a1.q[1] = *(const uint4*)(hstage + lm * 64 + 48 + 8 * g);
      v8f nacc[4] = {};
      #pragma unroll
      for (int t = 0; t < 4; ++t) {
        const v16h bf0 = *(const v16h*)(lds + base + t * 512 + lane * 16);
        const v16h bf1 = *(const v16h*)(lds + base + (4 + t) * 512 + lane * 16);
        nacc[t] = wmma_f16(a0.v, bf0, nacc[t]);
        nacc[t] = wmma_f16(a1.v, bf1, nacc[t]);
      }
      #pragma unroll
      for (int t = 0; t < 4; ++t) {
        const float bb = (layer == 0) ? bb2[t] : bb3[t];
        #pragma unroll
        for (int r = 0; r < 8; ++r)
          hstage[(r + 8 * g) * 64 + t * 16 + lm] = (_Float16)fmaxf(nacc[t][r] + bb, 0.f);
      }
    }

    // ---- Layer 4: K = 64, no relu; scatter-add into msgbuf ----
    {
      AFrag a0, a1;
      a0.q[0] = *(const uint4*)(hstage + lm * 64 + 8 * g);
      a0.q[1] = *(const uint4*)(hstage + lm * 64 + 16 + 8 * g);
      a1.q[0] = *(const uint4*)(hstage + lm * 64 + 32 + 8 * g);
      a1.q[1] = *(const uint4*)(hstage + lm * 64 + 48 + 8 * g);
      v8f m4[4] = {};
      #pragma unroll
      for (int t = 0; t < 4; ++t) {
        const v16h bf0 = *(const v16h*)(lds + 18432 + t * 512 + lane * 16);
        const v16h bf1 = *(const v16h*)(lds + 18432 + (4 + t) * 512 + lane * 16);
        m4[t] = wmma_f16(a0.v, bf0, m4[t]);
        m4[t] = wmma_f16(a1.v, bf1, m4[t]);
      }
      // Row M's dst index is already in lane M (loaded during the gather):
      // fetch via wave shuffle instead of 8 dependent global re-loads per lane.
      long long rowOff[8];
      #pragma unroll
      for (int r = 0; r < 8; ++r) {
        const int drow = __shfl(dst, r + 8 * g, 32);
        rowOff[r] = (long long)drow * 256 + typeOff + lm;
      }
      if (e0 + 16 <= E_) {              // full tile: no per-element guards
        #pragma unroll
        for (int t = 0; t < 4; ++t) {
          #pragma unroll
          for (int r = 0; r < 8; ++r) {
            const float v = m4[t][r] + bb4[t];
            __hip_atomic_fetch_add(&msgbuf[rowOff[r] + t * 16],
                                   v, __ATOMIC_RELAXED, __HIP_MEMORY_SCOPE_AGENT);
          }
        }
      } else {                          // tail tile
        #pragma unroll
        for (int t = 0; t < 4; ++t) {
          #pragma unroll
          for (int r = 0; r < 8; ++r) {
            if (e0 + r + 8 * g < E_) {
              const float v = m4[t][r] + bb4[t];
              __hip_atomic_fetch_add(&msgbuf[rowOff[r] + t * 16],
                                     v, __ATOMIC_RELAXED, __HIP_MEMORY_SCOPE_AGENT);
            }
          }
        }
      }
    }
  }
}

// ---------------------------------------------------------------------------
// LSTM cell: block = 16 nodes; wave w owns gate columns [32w, 32w+32).
// A rows = nodes, K = 384 ( x[0,64) | msgs[64,320) | rh[320,384) ).
// B fragments loaded straight from the pre-swizzled global buffer (L2-resident).
// ---------------------------------------------------------------------------
__global__ __launch_bounds__(256) void lstm_kernel(
    const float* __restrict__ xbuf, const float* __restrict__ msgbuf,
    const _Float16* rh_in, const _Float16* __restrict__ lfrag,
    float* __restrict__ cbuf, _Float16* hg_out, _Float16* rh_out, int N_)
{
  __shared__ __align__(16) float gates[16 * 256];
  const int wave = threadIdx.x >> 5;
  const int lane = threadIdx.x & 31;
  const int g = lane >> 4, lm = lane & 15;
  const int nBase = blockIdx.x * 16;
  int node = nBase + lm;
  if (node >= N_) node = N_ - 1;
  const float*    xr = xbuf   + (long long)node * 64;
  const float*    mr = msgbuf + (long long)node * 256;
  const _Float16* hr = rh_in  + (long long)node * 64;

  v8f acc0 = {}, acc1 = {};
  #pragma unroll
  for (int kc = 0; kc < 12; ++kc) {
    AFrag a;
    if (kc < 10) {  // f32 sources: x (kc<2) or msgs; two 8-float K-runs
      const float* srcp = (kc < 2) ? (xr + kc * 32) : (mr + (kc * 32 - 64));
      const float4 p0 = *(const float4*)(srcp + 8 * g);
      const float4 p1 = *(const float4*)(srcp + 8 * g + 4);
      const float4 p2 = *(const float4*)(srcp + 16 + 8 * g);
      const float4 p3 = *(const float4*)(srcp + 16 + 8 * g + 4);
      a.u[0] = pack2(p0.x, p0.y); a.u[1] = pack2(p0.z, p0.w);
      a.u[2] = pack2(p1.x, p1.y); a.u[3] = pack2(p1.z, p1.w);
      a.u[4] = pack2(p2.x, p2.y); a.u[5] = pack2(p2.z, p2.w);
      a.u[6] = pack2(p3.x, p3.y); a.u[7] = pack2(p3.z, p3.w);
    } else {        // f16 source: rh
      const _Float16* srcp = hr + (kc * 32 - 320);
      a.q[0] = *(const uint4*)(srcp + 8 * g);
      a.q[1] = *(const uint4*)(srcp + 16 + 8 * g);
    }
    const v16h b0 = *(const v16h*)(lfrag + (long long)(kc * 16 + 2 * wave)     * 512 + lane * 16);
    const v16h b1 = *(const v16h*)(lfrag + (long long)(kc * 16 + 2 * wave + 1) * 512 + lane * 16);
    acc0 = wmma_f16(a.v, b0, acc0);
    acc1 = wmma_f16(a.v, b1, acc1);
  }
  #pragma unroll
  for (int r = 0; r < 8; ++r) {
    gates[(r + 8 * g) * 256 + wave * 32 + lm]      = acc0[r];
    gates[(r + 8 * g) * 256 + wave * 32 + 16 + lm] = acc1[r];
  }
  __syncthreads();

  const int j  = threadIdx.x & 63;
  const int mb = threadIdx.x >> 6;
  #pragma unroll
  for (int k = 0; k < 4; ++k) {
    const int m = mb * 4 + k;
    const int n = nBase + m;
    if (n < N_) {
      const float gi = gates[m * 256 + j];
      const float gf = gates[m * 256 + 64 + j];
      const float gg = gates[m * 256 + 128 + j];
      const float go = gates[m * 256 + 192 + j];
      const float cold = cbuf[(long long)n * 64 + j];
      const float c = sigmoidf(gf) * cold + sigmoidf(gi) * tanhf(gg);
      const float h = sigmoidf(go) * tanhf(c);
      cbuf[(long long)n * 64 + j] = c;
      hg_out[(long long)n * 64 + j] = (_Float16)h;
      rh_out[(long long)n * 64 + j] = (_Float16)h;
    }
  }
}

__global__ __launch_bounds__(256) void score_kernel(
    const _Float16* __restrict__ h, const float* __restrict__ sw,
    float* __restrict__ out, int N_)
{
  const int n = blockIdx.x * 256 + threadIdx.x;
  if (n >= N_) return;
  float s = 0.f;
  #pragma unroll
  for (int j = 0; j < 64; ++j) s += (float)h[(long long)n * 64 + j] * sw[j];
  out[n] = s;
}

// ---------------------------------------------------------------------------
extern "C" void kernel_launch(void* const* d_in, const int* in_sizes, int n_in,
                              void* d_out, int out_size, void* d_ws, size_t ws_size,
                              hipStream_t stream) {
  const int N_ = in_sizes[0];
  const int E_ = in_sizes[1] / 2;

  const int* cell_idx = (const int*)d_in[0];
  const int* edgeptr[4] = {(const int*)d_in[1], (const int*)d_in[2],
                           (const int*)d_in[3], (const int*)d_in[4]};
  const float* embed = (const float*)d_in[5];
  const float *W1[4], *B1[4], *W2[4], *B2[4], *W3[4], *B3[4], *W4[4], *B4[4], *EE[4];
  for (int t = 0; t < 4; ++t) {
    const int b = 6 + 9 * t;
    W1[t] = (const float*)d_in[b + 0]; B1[t] = (const float*)d_in[b + 1];
    W2[t] = (const float*)d_in[b + 2]; B2[t] = (const float*)d_in[b + 3];
    W3[t] = (const float*)d_in[b + 4]; B3[t] = (const float*)d_in[b + 5];
    W4[t] = (const float*)d_in[b + 6]; B4[t] = (const float*)d_in[b + 7];
    EE[t] = (const float*)d_in[b + 8];
  }
  const float* w_ih = (const float*)d_in[42];
  const float* w_hh = (const float*)d_in[43];
  const float* sw   = (const float*)d_in[44];

  // Workspace carve-out (256B-aligned regions), ~58 MB total.
  char* ws = (char*)d_ws;
  size_t off = 0;
  auto carve = [&](size_t bytes) {
    void* p = ws + off;
    off = (off + bytes + 255) & ~(size_t)255;
    return p;
  };
  float*     xbuf   = (float*)    carve((size_t)N_ * 64 * 4);
  _Float16*  hg     = (_Float16*) carve((size_t)N_ * 64 * 2);
  _Float16*  rh     = (_Float16*) carve((size_t)N_ * 64 * 2);
  float*     cbuf   = (float*)    carve((size_t)N_ * 64 * 4);
  float*     msgbuf = (float*)    carve((size_t)N_ * 256 * 4);
  _Float16*  mfrag[4];
  for (int t = 0; t < 4; ++t) mfrag[t] = (_Float16*)carve(22528 * 2);
  _Float16*  lfrag  = (_Float16*) carve(98304 * 2);

  for (int t = 0; t < 4; ++t)
    prep_mlp_frag<<<88, 256, 0, stream>>>(W1[t], W2[t], W3[t], W4[t], mfrag[t]);
  prep_lstm_frag<<<384, 256, 0, stream>>>(w_ih, w_hh, lfrag);

  const long long nh = (long long)N_ * 64;
  init_state<<<(unsigned)((nh + 255) / 256), 256, 0, stream>>>(
      cell_idx, embed, xbuf, hg, rh, cbuf, N_);

  float* out = (float*)d_out;
  // Persistent grid for the edge MLP: enough blocks to fill the chip, capped so
  // each block amortizes its 45KB LDS weight load over many 16-edge tiles.
  const int numTiles = (E_ + 15) / 16;
  unsigned edgeBlocks = (unsigned)((numTiles + 7) / 8);
  if (edgeBlocks > 768u) edgeBlocks = 768u;
  const long long nm = (long long)N_ * 256;
  for (int s = 0; s < 4; ++s) {
    zero_msg<<<(unsigned)((nm + 255) / 256), 256, 0, stream>>>(msgbuf, N_);
    for (int t = 0; t < 4; ++t)
      edge_mlp_kernel<<<edgeBlocks, 256, 0, stream>>>(
          hg, edgeptr[t], mfrag[t], B1[t], B2[t], B3[t], B4[t], EE[t],
          msgbuf, t * 64, E_, N_);
    lstm_kernel<<<(unsigned)((N_ + 15) / 16), 256, 0, stream>>>(
        xbuf, msgbuf, rh, lfrag, cbuf, hg, rh, N_);
    score_kernel<<<(unsigned)((N_ + 255) / 256), 256, 0, stream>>>(
        hg, sw, out + (long long)s * N_, N_);
  }
}
